// MultiHeadAttention_38766374814133
// MI455X (gfx1250) — matrix operand order repaired and checked by reading, not compiled, not run on hardware
//
#include <hip/hip_runtime.h>

// ---------------------------------------------------------------------------
// MI455X (gfx1250, wave32) fused MHA: bf16 WMMA (v_wmma_f32_16x16x32_bf16)
// with fp32 accumulation + CDNA5 async/TDM LDS staging where available.
// ---------------------------------------------------------------------------

typedef __attribute__((ext_vector_type(16))) __bf16 v16bf;
typedef __attribute__((ext_vector_type(8)))  float  v8f;
typedef __attribute__((ext_vector_type(4)))  unsigned int v4u;
typedef __attribute__((ext_vector_type(8)))  int    v8i_t;
typedef __attribute__((ext_vector_type(4)))  int    v4i_t;

#define EMBED 1280
#define SEQ   1024
#define NHEAD 20
#define HDIM  64

// ----- CDNA5 feature probes -------------------------------------------------
#if __has_builtin(__builtin_amdgcn_global_load_async_to_lds_b128)
#define HAVE_ASYNC_LDS 1
#else
#define HAVE_ASYNC_LDS 0
#endif
#if __has_builtin(__builtin_amdgcn_tensor_load_to_lds) && \
    __has_builtin(__builtin_amdgcn_s_wait_tensorcnt)
#define HAVE_TDM 1
#else
#define HAVE_TDM 0
#endif

// Generic->LDS: LDS generic addresses keep the byte offset in addr[31:0]
// (ISA 10.2 aperture mapping), so truncation yields the AS3 offset.
static __device__ inline unsigned lds_offset_u32(const void* p) {
  return (unsigned)(unsigned long long)p;
}

// 16-byte global->LDS copy: async (ASYNCcnt) when available, sync fallback.
static __device__ inline void cp_b128(const unsigned int* g, unsigned int* l) {
#if HAVE_ASYNC_LDS
  typedef __attribute__((address_space(3))) v4i_t* lds_v4p;
  __builtin_amdgcn_global_load_async_to_lds_b128(
      (v4i_t*)g,
      (lds_v4p)(unsigned long long)lds_offset_u32(l),
      0, 0);
#else
  *reinterpret_cast<uint4*>(l) = *reinterpret_cast<const uint4*>(g);
#endif
}
static __device__ inline void wait_async_copies() {
#if HAVE_ASYNC_LDS
#if __has_builtin(__builtin_amdgcn_s_wait_asynccnt)
  __builtin_amdgcn_s_wait_asynccnt(0);
#else
  asm volatile("s_wait_asynccnt 0x0" ::: "memory");
#endif
#endif
}

// ----- bf16 helpers ---------------------------------------------------------
union FragBF {
  v16bf v;
  unsigned int u[8];
};

static __device__ inline unsigned short f2bf_bits(float f) {
  unsigned u = __builtin_bit_cast(unsigned, f);
  unsigned r = (u + 0x7FFFu + ((u >> 16) & 1u)) >> 16;
  return (unsigned short)r;
}
static __device__ inline unsigned int pack2(float a, float b) {
  return (unsigned int)f2bf_bits(a) | ((unsigned int)f2bf_bits(b) << 16);
}
static __device__ inline v8f vzero8() {
  v8f z;
  #pragma unroll
  for (int i = 0; i < 8; ++i) z[i] = 0.0f;
  return z;
}
static __device__ inline v8f wmma_bf16(const FragBF& a, const FragBF& b, v8f c) {
  return __builtin_amdgcn_wmma_f32_16x16x32_bf16(
      false, a.v, false, b.v, (short)0, c, false, false);
}

// ---------------------------------------------------------------------------
// Kernel 1: QKV projection + RoPE.  grid=(64,20,3) block=256 (8 waves)
// ---------------------------------------------------------------------------
__global__ __launch_bounds__(256)
void mha_qkv_rope_kernel(const float* __restrict__ x,
                         const float* __restrict__ Wq,
                         const float* __restrict__ Wk,
                         const float* __restrict__ Wv,
                         unsigned short* __restrict__ qb,
                         unsigned short* __restrict__ kb,
                         unsigned short* __restrict__ vb) {
  const int mode = blockIdx.z;
  const float* W = (mode == 0) ? Wq : ((mode == 1) ? Wk : Wv);
  unsigned short* outp = (mode == 0) ? qb : ((mode == 1) ? kb : vb);

  const int m0 = blockIdx.x * 128;
  const int n0 = blockIdx.y * 64;
  const int head = blockIdx.y;

  const int t  = threadIdx.x;
  const int w  = t >> 5;
  const int l  = t & 31;
  const int lm = l & 15;
  const int h  = l >> 4;

  __shared__ __align__(16) unsigned int Als[128 * 16]; // A 128x32 bf16  8 KB
  __shared__ __align__(16) unsigned int Bls[64 * 16];  // Bt 64n x 32k   4 KB

  v8f acc[4];
  #pragma unroll
  for (int nt = 0; nt < 4; ++nt) acc[nt] = vzero8();

  const float2* x2 = reinterpret_cast<const float2*>(x);

  for (int k0 = 0; k0 < EMBED; k0 += 32) {
    __syncthreads();
    // Stage A (fp32 -> bf16 conversion, VALU path)
    #pragma unroll
    for (int i = 0; i < 8; ++i) {
      int e = t + i * 256;
      int row = e >> 4, cp = e & 15;
      float2 xv = x2[((size_t)(m0 + row) * EMBED + k0) / 2 + cp];
      Als[row * 16 + cp] = pack2(xv.x, xv.y);
    }
    // Stage B transposed (fp32 -> bf16)
    #pragma unroll
    for (int i = 0; i < 4; ++i) {
      int e = t + i * 256;
      int kp = e >> 6, n = e & 63;
      float b0 = W[(size_t)(k0 + 2 * kp) * EMBED + n0 + n];
      float b1 = W[(size_t)(k0 + 2 * kp + 1) * EMBED + n0 + n];
      Bls[n * 16 + kp] = pack2(b0, b1);
    }
    if (k0 + 32 < EMBED)
      __builtin_prefetch(&x[(size_t)(m0 + (t & 127)) * EMBED + k0 + 32], 0, 1);
    __syncthreads();

    FragBF af;
    const int mrow = w * 16 + lm;
    #pragma unroll
    for (int i = 0; i < 8; ++i) {
      int kk = h * 16 + i * 2;
      af.u[i] = Als[mrow * 16 + (kk >> 1)];
    }
    #pragma unroll
    for (int nt = 0; nt < 4; ++nt) {
      FragBF bfg;
      int n = nt * 16 + lm;
      #pragma unroll
      for (int r = 0; r < 8; ++r) bfg.u[r] = Bls[n * 16 + h * 8 + r];
      acc[nt] = wmma_bf16(af, bfg, acc[nt]);
    }
  }

  const int row_base = m0 + w * 16;
  const int b = m0 / SEQ;

  if (mode < 2) {
    #pragma unroll
    for (int j = 0; j < 2; ++j) {
      int dp = j * 16 + lm;
      float inv_ts = __expf(-(float)dp * 0.28782313662425f); // ln(1e4)/32
      #pragma unroll
      for (int r = 0; r < 8; ++r) {
        int s = (row_base + r + 8 * h) & (SEQ - 1);
        float ang = (float)s * inv_ts;
        float sn = __sinf(ang), cs = __cosf(ang);
        float x0 = acc[j][r], x1 = acc[j + 2][r];
        acc[j][r]     = x0 * cs - x1 * sn;
        acc[j + 2][r] = x1 * cs + x0 * sn;
      }
    }
    #pragma unroll
    for (int nt = 0; nt < 4; ++nt) {
      int d = nt * 16 + lm;
      #pragma unroll
      for (int r = 0; r < 8; ++r) {
        int s = (row_base + r + 8 * h) & (SEQ - 1);
        size_t off = ((size_t)(b * NHEAD + head) * SEQ + s) * HDIM + d;
        outp[off] = f2bf_bits(acc[nt][r]);
      }
    }
  } else {
    #pragma unroll
    for (int nt = 0; nt < 4; ++nt) {
      int d = nt * 16 + lm;
      #pragma unroll
      for (int r = 0; r < 8; ++r) {
        int s = (row_base + r + 8 * h) & (SEQ - 1);
        size_t off = ((size_t)(b * NHEAD + head) * HDIM + d) * SEQ + s;
        outp[off] = f2bf_bits(acc[nt][r]);
      }
    }
  }
}

// ---------------------------------------------------------------------------
// Kernel 2: flash attention.  grid=(16,20,8) block=128 (4 waves)
// ---------------------------------------------------------------------------
__global__ __launch_bounds__(128)
void mha_flash_attn_kernel(const unsigned short* __restrict__ qb,
                           const unsigned short* __restrict__ kb,
                           const unsigned short* __restrict__ vb,
                           unsigned short* __restrict__ ctx) {
  const int q0   = blockIdx.x * 64;
  const int head = blockIdx.y;
  const int b    = blockIdx.z;

  const int t  = threadIdx.x;
  const int w  = t >> 5;
  const int l  = t & 31;
  const int lm = l & 15;
  const int h  = l >> 4;

  __shared__ __align__(16) unsigned int  Kls[32 * 32];     // 4 KB
  __shared__ __align__(16) unsigned int  Vls[64 * 16];     // 4 KB
  __shared__ __align__(16) unsigned short Pls[4][16 * 32]; // 4 KB

  const unsigned int* qbu = reinterpret_cast<const unsigned int*>(qb);
  const unsigned int* kbu = reinterpret_cast<const unsigned int*>(kb);
  const unsigned int* vbu = reinterpret_cast<const unsigned int*>(vb);

  const size_t bh = (size_t)(b * NHEAD + head);

  FragBF qf[2];
  {
    const int sq = q0 + w * 16 + lm;
    const size_t qrow = (bh * SEQ + sq) * (HDIM / 2);
    #pragma unroll
    for (int ks = 0; ks < 2; ++ks) {
      #pragma unroll
      for (int i = 0; i < 8; ++i) {
        int kk = h * 16 + i * 2;
        qf[ks].u[i] = qbu[qrow + ks * 16 + (kk >> 1)];
      }
    }
  }

  v8f O[4];
  #pragma unroll
  for (int nt = 0; nt < 4; ++nt) O[nt] = vzero8();
  float mrow[8], lrow[8];
  #pragma unroll
  for (int r = 0; r < 8; ++r) { mrow[r] = -1e30f; lrow[r] = 0.0f; }

  for (int key0 = 0; key0 < SEQ; key0 += 32) {
    __syncthreads();
    const unsigned int* ksrc = kbu + (bh * SEQ + key0) * (HDIM / 2);

#if HAVE_TDM
    // K tile via Tensor Data Mover: contiguous 1-D tile of 2048 bf16 (4 KB).
    if (w == 0) {
      unsigned lds_base = lds_offset_u32(&Kls[0]);
      unsigned long long ga = (unsigned long long)(const void*)ksrc;
      v4u g0;
      g0[0] = 1u;                                  // count=1, user mode
      g0[1] = lds_base;                            // lds_addr
      g0[2] = (unsigned)ga;                        // global_addr[31:0]
      g0[3] = (unsigned)((ga >> 32) & 0x1FFFFFFu)  // global_addr[56:32]
              | (2u << 30);                        // type=2 (image)
      v8i_t g1;
      g1[0] = (int)(1u << 16);     // workgroup_mask=0, data_size=1 (2 bytes)
      g1[1] = (int)(2048u << 16);  // tensor_dim0[15:0] -> 2048 elements
      g1[2] = 0;                   // tensor_dim0 hi, tensor_dim1 lo
      g1[3] = (int)(2048u << 16);  // tensor_dim1 hi=0, tile_dim0=2048
      g1[4] = 0;                   // tile_dim1=0 (1-D), tile_dim2=0
      g1[5] = 2048;                // tensor_dim0_stride lo
      g1[6] = 0;
      g1[7] = 0;
      v4i_t g2; g2[0] = 0; g2[1] = 0; g2[2] = 0; g2[3] = 0;
      v4i_t g3 = g2;
      v8i_t gx;
      #pragma unroll
      for (int i = 0; i < 8; ++i) gx[i] = 0;
      __builtin_amdgcn_tensor_load_to_lds(g0, g1, g2, g3, gx, 0);
      __builtin_amdgcn_s_wait_tensorcnt(0);
    }
#else
    #pragma unroll
    for (int i = 0; i < 2; ++i) {
      int c = t + i * 128;                 // 256 x 16B chunks
      cp_b128(ksrc + c * 4, &Kls[c * 4]);
    }
#endif
    // V^T tile: 64 rows (d) x 64 B, async 16B chunks
    #pragma unroll
    for (int i = 0; i < 2; ++i) {
      int c = t + i * 128;
      int d = c >> 2, q4 = c & 3;
      cp_b128(vbu + (bh * HDIM + d) * (SEQ / 2) + (key0 >> 1) + q4 * 4,
              &Vls[d * 16 + q4 * 4]);
    }
    wait_async_copies();
    if (key0 + 32 < SEQ)
      __builtin_prefetch(&kbu[(bh * SEQ + key0 + 32) * (HDIM / 2) + t], 0, 1);
    __syncthreads();

    // Logits: 16 q x 32 keys (two C tiles), K=64 in two steps
    v8f c[2];
    c[0] = vzero8(); c[1] = vzero8();
    #pragma unroll
    for (int nk = 0; nk < 2; ++nk) {
      int key = nk * 16 + lm;
      #pragma unroll
      for (int ks = 0; ks < 2; ++ks) {
        FragBF kf;
        #pragma unroll
        for (int r = 0; r < 8; ++r)
          kf.u[r] = Kls[key * 32 + ks * 16 + h * 8 + r];
        c[nk] = wmma_bf16(qf[ks], kf, c[nk]);
      }
    }

    // Online softmax (rows span the 16 lanes of each half-wave)
    float corr[8];
    #pragma unroll
    for (int r = 0; r < 8; ++r) {
      float v0 = c[0][r] * 0.125f;
      float v1 = c[1][r] * 0.125f;
      float mx = fmaxf(v0, v1);
      #pragma unroll
      for (int o = 1; o < 16; o <<= 1) mx = fmaxf(mx, __shfl_xor(mx, o, 32));
      float mnew = fmaxf(mrow[r], mx);
      float p0 = __expf(v0 - mnew);
      float p1 = __expf(v1 - mnew);
      float ssum = p0 + p1;
      #pragma unroll
      for (int o = 1; o < 16; o <<= 1) ssum += __shfl_xor(ssum, o, 32);
      float cr = __expf(mrow[r] - mnew);
      lrow[r] = lrow[r] * cr + ssum;
      mrow[r] = mnew;
      corr[r] = cr;
      int rowm = r + 8 * h;
      Pls[w][rowm * 32 + lm]      = f2bf_bits(p0);
      Pls[w][rowm * 32 + 16 + lm] = f2bf_bits(p1);
    }
    #pragma unroll
    for (int nt = 0; nt < 4; ++nt)
      #pragma unroll
      for (int r = 0; r < 8; ++r) O[nt][r] *= corr[r];

    // P: C layout -> A fragment via per-wave LDS buffer (no barrier needed)
    const unsigned int* Pu = reinterpret_cast<const unsigned int*>(&Pls[w][0]);
    FragBF pf;
    #pragma unroll
    for (int i = 0; i < 8; ++i) {
      int kk = h * 16 + i * 2;
      pf.u[i] = Pu[lm * 16 + (kk >> 1)];
    }
    #pragma unroll
    for (int nt = 0; nt < 4; ++nt) {
      FragBF vf;
      int nd = nt * 16 + lm;
      #pragma unroll
      for (int r = 0; r < 8; ++r) vf.u[r] = Vls[nd * 16 + h * 8 + r];
      O[nt] = wmma_bf16(pf, vf, O[nt]);
    }
  }

  #pragma unroll
  for (int nt = 0; nt < 4; ++nt) {
    int d = nt * 16 + lm;
    #pragma unroll
    for (int r = 0; r < 8; ++r) {
      int s = q0 + w * 16 + r + 8 * h;
      float val = O[nt][r] / lrow[r];
      ctx[((size_t)b * SEQ + s) * EMBED + head * HDIM + d] = f2bf_bits(val);
    }
  }
}

// ---------------------------------------------------------------------------
// Kernel 3: output projection.  grid=(64,20) block=256
// ---------------------------------------------------------------------------
__global__ __launch_bounds__(256)
void mha_out_proj_kernel(const unsigned short* __restrict__ ctx,
                         const float* __restrict__ Wo,
                         float* __restrict__ out) {
  const int m0 = blockIdx.x * 128;
  const int n0 = blockIdx.y * 64;

  const int t  = threadIdx.x;
  const int w  = t >> 5;
  const int l  = t & 31;
  const int lm = l & 15;
  const int h  = l >> 4;

  __shared__ __align__(16) unsigned int Als[128 * 16];
  __shared__ __align__(16) unsigned int Bls[64 * 16];

  v8f acc[4];
  #pragma unroll
  for (int nt = 0; nt < 4; ++nt) acc[nt] = vzero8();

  const unsigned int* cu = reinterpret_cast<const unsigned int*>(ctx);

  for (int k0 = 0; k0 < EMBED; k0 += 32) {
    __syncthreads();
    // Stage A (already bf16): async 16B chunks; 128 rows x 64 B
    #pragma unroll
    for (int i = 0; i < 2; ++i) {
      int c = t + i * 256;                 // 512 chunks
      int row = c >> 2, q4 = c & 3;
      cp_b128(cu + (size_t)(m0 + row) * (EMBED / 2) + (k0 >> 1) + q4 * 4,
              &Als[row * 16 + q4 * 4]);
    }
    // Stage B transposed (fp32 -> bf16)
    #pragma unroll
    for (int i = 0; i < 4; ++i) {
      int e = t + i * 256;
      int kp = e >> 6, n = e & 63;
      float b0 = Wo[(size_t)(k0 + 2 * kp) * EMBED + n0 + n];
      float b1 = Wo[(size_t)(k0 + 2 * kp + 1) * EMBED + n0 + n];
      Bls[n * 16 + kp] = pack2(b0, b1);
    }
    wait_async_copies();
    __syncthreads();

    FragBF af;
    const int mrow = w * 16 + lm;
    #pragma unroll
    for (int i = 0; i < 8; ++i) {
      int kk = h * 16 + i * 2;
      af.u[i] = Als[mrow * 16 + (kk >> 1)];
    }
    #pragma unroll
    for (int nt = 0; nt < 4; ++nt) {
      FragBF bfg;
      int n = nt * 16 + lm;
      #pragma unroll
      for (int r = 0; r < 8; ++r) bfg.u[r] = Bls[n * 16 + h * 8 + r];
      acc[nt] = wmma_bf16(af, bfg, acc[nt]);
    }
  }

  #pragma unroll
  for (int nt = 0; nt < 4; ++nt) {
    int n = n0 + nt * 16 + lm;
    #pragma unroll
    for (int r = 0; r < 8; ++r) {
      int row = m0 + w * 16 + r + 8 * h;
      out[(size_t)row * EMBED + n] = acc[nt][r];
    }
  }
}

// ---------------------------------------------------------------------------
// Launch
// ---------------------------------------------------------------------------
extern "C" void kernel_launch(void* const* d_in, const int* in_sizes, int n_in,
                              void* d_out, int out_size, void* d_ws, size_t ws_size,
                              hipStream_t stream) {
  (void)in_sizes; (void)n_in; (void)out_size; (void)ws_size;
  const float* x  = (const float*)d_in[0];
  const float* Wq = (const float*)d_in[1];
  const float* Wk = (const float*)d_in[2];
  const float* Wv = (const float*)d_in[3];
  const float* Wo = (const float*)d_in[4];
  float* out = (float*)d_out;

  const size_t QK_BYTES = (size_t)8 * NHEAD * SEQ * HDIM * 2; // 20 MB each
  char* ws = (char*)d_ws;
  unsigned short* qb  = (unsigned short*)(ws);
  unsigned short* kb  = (unsigned short*)(ws + QK_BYTES);
  unsigned short* vb  = (unsigned short*)(ws + 2 * QK_BYTES);
  unsigned short* ctx = (unsigned short*)(ws + 3 * QK_BYTES);

  mha_qkv_rope_kernel<<<dim3(64, 20, 3), 256, 0, stream>>>(x, Wq, Wk, Wv, qb, kb, vb);
  mha_flash_attn_kernel<<<dim3(16, 20, 8), 128, 0, stream>>>(qb, kb, vb, ctx);
  mha_out_proj_kernel<<<dim3(64, 20), 256, 0, stream>>>(ctx, Wo, out);
}
